// MotionGPT_72670846648457
// MI455X (gfx1250) — matrix operand order repaired and checked by reading, not compiled, not run on hardware
//
#include <hip/hip_runtime.h>

// ---------------------------------------------------------------------------
// MotionGPT transformer forward, CDNA5 (gfx1250) WMMA bf16 implementation.
// B=2, T=64, V=32 -> S=2048 tokens, E=512, H=8 (dh=64), DFF=2048, L=6.
// Matmuls via v_wmma_f32_16x16x32_bf16; GEMM tiles staged in LDS via the
// Tensor Data Mover (tensor_load_to_lds, TENSORcnt) with double buffering.
// Wave tile 64x64 (16 WMMA per K-slice per wave, 1:1 WMMA:ds_load ratio).
// ---------------------------------------------------------------------------

typedef __attribute__((ext_vector_type(16))) __bf16    v16bf;
typedef __attribute__((ext_vector_type(8)))  __bf16    v8bf;
typedef __attribute__((ext_vector_type(8)))  float     v8f;
typedef __attribute__((ext_vector_type(8)))  unsigned  v8u;
typedef __attribute__((ext_vector_type(4)))  unsigned  v4u;
typedef __attribute__((ext_vector_type(8)))  int       v8i;
typedef __attribute__((ext_vector_type(4)))  int       v4i;

static constexpr int E_   = 512;
static constexpr int H_   = 8;
static constexpr int L_   = 6;
static constexpr int DFF_ = 2048;
static constexpr int S_   = 2048;    // T*V
static constexpr int B_   = 2;
static constexpr int DH_  = 64;      // E/H
static constexpr int LQK  = 1024;    // q|k buffer row stride
static constexpr int MTOK = B_ * S_; // 4096 rows

#if __has_builtin(__builtin_amdgcn_tensor_load_to_lds) && \
    __has_builtin(__builtin_amdgcn_s_wait_tensorcnt)
#define USE_TDM 1
#else
#define USE_TDM 0
#endif

__device__ __forceinline__ unsigned short bfbits(float f) {
  unsigned u = __builtin_bit_cast(unsigned, f);
  u += 0x7FFFu + ((u >> 16) & 1u);               // round-to-nearest-even
  return (unsigned short)(u >> 16);
}
__device__ __forceinline__ __bf16 f2bf(float f) {
  return __builtin_bit_cast(__bf16, bfbits(f));
}

__device__ __forceinline__ v16bf ld16(const __bf16* plo, const __bf16* phi) {
  v8bf lo = *(const v8bf*)plo;
  v8bf hi = *(const v8bf*)phi;
  v16bf r;
#pragma unroll
  for (int i = 0; i < 8; ++i) { r[i] = lo[i]; r[i + 8] = hi[i]; }
  return r;
}

__device__ __forceinline__ v8f wmma_bf16(v16bf a, v16bf b, v8f c) {
  return __builtin_amdgcn_wmma_f32_16x16x32_bf16(false, a, false, b, (short)0, c,
                                                 false, false);
}

#if USE_TDM
// --- Tensor Data Mover: load a (32 x tile_d1) bf16 tile to LDS -------------
// D# per CDNA5 ISA ch.8: group0 {count=1, lds_addr, global_addr, type=2},
// group1 {data_size=2B, tensor dims, tile dims, dim0 stride}.
__device__ __forceinline__ void tdm_load_tile(unsigned lds_off,
                                              const __bf16* gsrc,
                                              unsigned stride_elems,
                                              unsigned tens_d0,
                                              unsigned tens_d1,
                                              unsigned tile_d1) {
  unsigned long long ga = (unsigned long long)(const void*)gsrc;
  v4u g0;
  g0[0] = 1u;                                          // count=1, user mode
  g0[1] = lds_off;                                     // LDS byte address
  g0[2] = (unsigned)ga;                                // global_addr[31:0]
  g0[3] = (unsigned)((ga >> 32) & 0x01FFFFFFu) | (2u << 30); // addr[56:32]|type=2
  v8i g1;
  g1[0] = (int)(1u << 16);                             // data_size = 1 (2 bytes)
  g1[1] = (int)((tens_d0 & 0xFFFFu) << 16);            // tensor_dim0[15:0]
  g1[2] = (int)((tens_d0 >> 16) | ((tens_d1 & 0xFFFFu) << 16));
  g1[3] = (int)((tens_d1 >> 16) | (32u << 16));        // tile_dim0 = 32
  g1[4] = (int)tile_d1;                                // tile_dim1
  g1[5] = (int)stride_elems;                           // tensor_dim0_stride
  g1[6] = 0;
  g1[7] = 0;
  v4i z4 = (v4i)0;
#if __clang_major__ >= 23
  v8i z8 = (v8i)0;
  __builtin_amdgcn_tensor_load_to_lds(g0, g1, z4, z4, z8, 0);
#else
  __builtin_amdgcn_tensor_load_to_lds(g0, g1, z4, z4, 0);
#endif
}
#endif

// ---------------------------------------------------------------------------
// fp32 -> bf16 conversion (weights, once per launch)
// ---------------------------------------------------------------------------
__global__ void cvt_kernel(const float* __restrict__ src,
                           __bf16* __restrict__ dst, int n) {
  int i = blockIdx.x * 256 + threadIdx.x;
  if (i < n) dst[i] = f2bf(src[i]);
}

// ---------------------------------------------------------------------------
// Embedding: x = code_emb[idx] + stage_emb + time_pos[t] + joint_pos[v]
// ---------------------------------------------------------------------------
__global__ void embed_kernel(const int* __restrict__ idx,
                             const float* __restrict__ ce,
                             const float* __restrict__ se,
                             const float* __restrict__ tp,
                             const float* __restrict__ jp,
                             float* __restrict__ x) {
  int tok = blockIdx.x;
  int s   = tok & (S_ - 1);
  int t   = s >> 5;
  int v   = s & 31;
  int code = idx[tok];
  int tid = threadIdx.x; // 0..127
  float4 a  = ((const float4*)(ce + (size_t)code * E_))[tid];
  float4 b4 = ((const float4*)se)[tid];
  float4 c4 = ((const float4*)(tp + (size_t)t * E_))[tid];
  float4 d4 = ((const float4*)(jp + (size_t)v * E_))[tid];
  float4 o;
  o.x = a.x + b4.x + c4.x + d4.x;
  o.y = a.y + b4.y + c4.y + d4.y;
  o.z = a.z + b4.z + c4.z + d4.z;
  o.w = a.w + b4.w + c4.w + d4.w;
  ((float4*)(x + (size_t)tok * E_))[tid] = o;
}

// ---------------------------------------------------------------------------
// LayerNorm over E=512; one block (128 threads) per token; output bf16
// ---------------------------------------------------------------------------
__global__ void ln_kernel(const float* __restrict__ x,
                          const float* __restrict__ g,
                          const float* __restrict__ bta,
                          __bf16* __restrict__ y) {
  __shared__ float sb[8];
  int tok = blockIdx.x;
  int tid = threadIdx.x;
  float4 v = ((const float4*)(x + (size_t)tok * E_))[tid];
  float vals[4] = {v.x, v.y, v.z, v.w};
  float s = 0.f, s2 = 0.f;
#pragma unroll
  for (int i = 0; i < 4; ++i) { s += vals[i]; s2 += vals[i] * vals[i]; }
#pragma unroll
  for (int off = 16; off; off >>= 1) {
    s  += __shfl_xor(s,  off, 32);
    s2 += __shfl_xor(s2, off, 32);
  }
  int wv = tid >> 5, lane = tid & 31;
  if (lane == 0) { sb[wv] = s; sb[4 + wv] = s2; }
  __syncthreads();
  s  = sb[0] + sb[1] + sb[2] + sb[3];
  s2 = sb[4] + sb[5] + sb[6] + sb[7];
  float mu  = s * (1.f / E_);
  float var = s2 * (1.f / E_) - mu * mu;
  float inv = rsqrtf(var + 1e-5f);
  float4 gv = ((const float4*)g)[tid];
  float4 bv = ((const float4*)bta)[tid];
  float gs[4] = {gv.x, gv.y, gv.z, gv.w};
  float bs[4] = {bv.x, bv.y, bv.z, bv.w};
  __bf16* yo = y + (size_t)tok * E_ + tid * 4;
#pragma unroll
  for (int i = 0; i < 4; ++i)
    yo[i] = f2bf((vals[i] - mu) * inv * gs[i] + bs[i]);
}

// ---------------------------------------------------------------------------
// WMMA GEMM: C[M,N] = A[M,K](bf16) @ W[N,K](bf16)^T + bias
// 256 thr = 8 waves (2 M-groups x 4 N-groups); block tile 128x256;
// wave tile 64x64 (4x4 frags, 16 WMMA per 32-K slice); K step 32.
// A/B K-slices staged in LDS, double buffered; fill via TDM (wave 0 issues
// tensor_load_to_lds, waits TENSORcnt) or cooperative copies as fallback.
// MODE 0: bf16 store | 1: exact GELU->bf16 | 2: fp32 residual += |
// MODE 3: qkv split (n<1024 -> QK buffer, n>=1024 -> transposed V buffer)
// ---------------------------------------------------------------------------
template <int MODE>
__global__ __launch_bounds__(256) void gemm_kernel(
    const __bf16* __restrict__ A, const __bf16* __restrict__ W,
    const float* __restrict__ bias, float* __restrict__ R,
    __bf16* __restrict__ O, __bf16* __restrict__ O2, int M, int N, int K) {
  __shared__ __attribute__((aligned(16))) __bf16 As[2][128 * 32];
  __shared__ __attribute__((aligned(16))) __bf16 Bs[2][256 * 32];

  const int lane = threadIdx.x & 31, wave = threadIdx.x >> 5;
  const int half = lane >> 4, l16 = lane & 15;
  const int wm = wave & 1, wn = wave >> 1;     // 2 x 4 wave grid
  const int m0b = blockIdx.y * 128;
  const int n0b = blockIdx.x * 256;

  v8f acc[4][4] = {};

#if USE_TDM
  auto fill = [&](int st, int k0) {
    if (wave == 0) {
      tdm_load_tile((unsigned)(unsigned long long)(void*)&As[st][0],
                    A + (size_t)m0b * K + k0, (unsigned)K, (unsigned)K,
                    (unsigned)M, 128u);
      tdm_load_tile((unsigned)(unsigned long long)(void*)&Bs[st][0],
                    W + (size_t)n0b * K + k0, (unsigned)K, (unsigned)K,
                    (unsigned)N, 256u);
    }
  };
#else
  auto fill = [&](int st, int k0) {
    int t = threadIdx.x;
    // A: 128 rows, one 32B half-row per thread
    const __bf16* sA = A + (size_t)(m0b + (t >> 1)) * K + k0 + (t & 1) * 16;
    v8bf* dA = (v8bf*)&As[st][(t >> 1) * 32 + (t & 1) * 16];
    dA[0] = *(const v8bf*)sA;
    dA[1] = *(const v8bf*)(sA + 8);
    // B: 256 rows, one full 64B row per thread
    const __bf16* sB = W + (size_t)(n0b + t) * K + k0;
    v8bf* dB = (v8bf*)&Bs[st][t * 32];
#pragma unroll
    for (int c = 0; c < 4; ++c) dB[c] = *(const v8bf*)(sB + 8 * c);
  };
#endif

  fill(0, 0);
  for (int k = 0; k < K; k += 32) {
    const int st = (k >> 5) & 1;
    const bool nxt = (k + 32) < K;
    if (nxt) fill(st ^ 1, k + 32);
#if USE_TDM
    if (wave == 0) {
      if (nxt) __builtin_amdgcn_s_wait_tensorcnt(2);
      else     __builtin_amdgcn_s_wait_tensorcnt(0);
    }
#endif
    __syncthreads();

    v16bf b[4];
#pragma unroll
    for (int j = 0; j < 4; ++j) {
      const __bf16* Bb = &Bs[st][(wn * 64 + j * 16 + l16) * 32 + half * 16];
      b[j] = ld16(Bb, Bb + 8);
    }
#pragma unroll
    for (int i = 0; i < 4; ++i) {
      const __bf16* Ab = &As[st][(wm * 64 + i * 16 + l16) * 32 + half * 16];
      v16bf a = ld16(Ab, Ab + 8);
      acc[i][0] = wmma_bf16(a, b[0], acc[i][0]);
      acc[i][1] = wmma_bf16(a, b[1], acc[i][1]);
      acc[i][2] = wmma_bf16(a, b[2], acc[i][2]);
      acc[i][3] = wmma_bf16(a, b[3], acc[i][3]);
    }
    __syncthreads();
  }

#pragma unroll
  for (int j = 0; j < 4; ++j) {
    int n = n0b + wn * 64 + j * 16 + l16;
    float bv = bias[n];
#pragma unroll
    for (int i = 0; i < 4; ++i) {
#pragma unroll
      for (int r = 0; r < 8; ++r) {
        int m = m0b + wm * 64 + i * 16 + r + 8 * half;
        float v = acc[i][j][r] + bv;
        if (MODE == 0) {
          O[(size_t)m * N + n] = f2bf(v);
        } else if (MODE == 1) {
          O[(size_t)m * N + n] =
              f2bf(0.5f * v * (1.f + erff(v * 0.70710678118654752f)));
        } else if (MODE == 2) {
          R[(size_t)m * N + n] += v;
        } else {               // MODE 3: qkv split
          if (n < 2 * E_) {
            O[(size_t)m * LQK + n] = f2bf(v);         // Q|K, stride 1024
          } else {
            int c  = n - 2 * E_;
            int hv = c >> 6, d = c & 63;
            int bb = m >> 11, ss = m & (S_ - 1);
            O2[(size_t)((bb * H_ + hv) * DH_ + d) * S_ + ss] = f2bf(v);
          }
        }
      }
    }
  }
}

// ---------------------------------------------------------------------------
// Flash attention, block-causal (frame = 32 tokens -> whole key-tiles kept).
// grid (S/128, B*H); 8 waves, wave owns 16 query rows (single frame).
// Row-owner softmax: scores staged in per-wave LDS, lane r owns row r
// (key halves split across half-waves, combined with one shfl_xor(16));
// P A-fragment assembled in registers via packed half-wave exchanges.
// ---------------------------------------------------------------------------
__global__ __launch_bounds__(256) void attn_kernel(
    const __bf16* __restrict__ qk, const __bf16* __restrict__ Vt,
    __bf16* __restrict__ out) {
  __shared__ __attribute__((aligned(16))) float sbuf[8][16 * 32];
  __shared__ __attribute__((aligned(16))) float abuf[8][16];

  const int wave = threadIdx.x >> 5, lane = threadIdx.x & 31;
  const int half = lane >> 4, l16 = lane & 15;
  const int bh = blockIdx.y, b = bh >> 3, h = bh & 7;
  const int q0 = blockIdx.x * 128 + wave * 16;
  const int kend = ((q0 >> 5) + 1) << 5;    // (frame+1)*32

  const __bf16* qb = qk + (size_t)(b * S_ + q0 + l16) * LQK + h * DH_ + half * 16;
  v16bf qf0 = ld16(qb, qb + 8);             // d 0..31
  v16bf qf1 = ld16(qb + 32, qb + 40);       // d 32..63

  v8f o[4] = {};
  float mrow = -1e30f, lrow = 0.f;

  for (int kb = 0; kb < kend; kb += 32) {
    v8f c0 = {}, c1 = {};
    {
      const __bf16* kp =
          qk + (size_t)(b * S_ + kb + l16) * LQK + E_ + h * DH_ + half * 16;
      c0 = wmma_bf16(qf0, ld16(kp, kp + 8), c0);
      c0 = wmma_bf16(qf1, ld16(kp + 32, kp + 40), c0);
    }
    {
      const __bf16* kp =
          qk + (size_t)(b * S_ + kb + 16 + l16) * LQK + E_ + h * DH_ + half * 16;
      c1 = wmma_bf16(qf0, ld16(kp, kp + 8), c1);
      c1 = wmma_bf16(qf1, ld16(kp + 32, kp + 40), c1);
    }

    // stash scaled scores (C layout -> row-major LDS)
    float* srow = sbuf[wave];
#pragma unroll
    for (int r = 0; r < 8; ++r) {
      int row = r + 8 * half;
      srow[row * 32 + l16]      = c0[r] * 0.125f;   // 1/sqrt(dh)
      srow[row * 32 + 16 + l16] = c1[r] * 0.125f;
    }
    asm volatile("s_wait_dscnt 0" ::: "memory");

    // row-owner phase: this lane owns row l16, keys half*16..half*16+15
    float sv[16];
    {
      const float4* sp = (const float4*)&srow[l16 * 32 + half * 16];
#pragma unroll
      for (int i = 0; i < 4; ++i) {
        float4 t = sp[i];
        sv[4 * i] = t.x; sv[4 * i + 1] = t.y;
        sv[4 * i + 2] = t.z; sv[4 * i + 3] = t.w;
      }
    }
    float tmax = sv[0];
#pragma unroll
    for (int i = 1; i < 16; ++i) tmax = fmaxf(tmax, sv[i]);
    tmax = fmaxf(tmax, __shfl_xor(tmax, 16, 32));
    float nm = fmaxf(mrow, tmax);
    float al = __expf(mrow - nm);
    float ps = 0.f;
    unsigned u[8];
#pragma unroll
    for (int i = 0; i < 8; ++i) {
      float e0 = __expf(sv[2 * i] - nm);
      float e1 = __expf(sv[2 * i + 1] - nm);
      ps += e0 + e1;
      u[i] = (unsigned)bfbits(e0) | ((unsigned)bfbits(e1) << 16);
    }
    ps += __shfl_xor(ps, 16, 32);
    lrow = lrow * al + ps;
    mrow = nm;
    if (half == 0) abuf[wave][l16] = al;

    // assemble P A-fragment in registers: exchange 4 packed uints with partner
    unsigned w[8];
#pragma unroll
    for (int i = 0; i < 4; ++i) {
      unsigned snd = half ? u[i] : u[4 + i];
      unsigned rcv = __shfl_xor(snd, 16, 32);
      if (half) { w[i] = rcv;  w[4 + i] = u[4 + i]; }
      else      { w[i] = u[i]; w[4 + i] = rcv; }
    }
    v8u wvv;
#pragma unroll
    for (int i = 0; i < 8; ++i) wvv[i] = w[i];
    v16bf pf = __builtin_bit_cast(v16bf, wvv);

    asm volatile("s_wait_dscnt 0" ::: "memory");
    float4 a0 = *(const float4*)&abuf[wave][8 * half];
    float4 a1 = *(const float4*)&abuf[wave][8 * half + 4];
    float alr[8] = {a0.x, a0.y, a0.z, a0.w, a1.x, a1.y, a1.z, a1.w};
#pragma unroll
    for (int j = 0; j < 4; ++j)
#pragma unroll
      for (int r = 0; r < 8; ++r) o[j][r] *= alr[r];

#pragma unroll
    for (int j = 0; j < 4; ++j) {
      const __bf16* vp =
          Vt + (size_t)(bh * DH_ + j * 16 + l16) * S_ + kb + half * 8;
      o[j] = wmma_bf16(pf, ld16(vp, vp + 16), o[j]);
    }
    asm volatile("" ::: "memory");
  }

  // normalize (1/l broadcast through LDS) and store concat-heads bf16
  if (half == 0) abuf[wave][l16] = 1.f / lrow;
  asm volatile("s_wait_dscnt 0" ::: "memory");
  float4 i0 = *(const float4*)&abuf[wave][8 * half];
  float4 i1 = *(const float4*)&abuf[wave][8 * half + 4];
  float inv[8] = {i0.x, i0.y, i0.z, i0.w, i1.x, i1.y, i1.z, i1.w};
#pragma unroll
  for (int j = 0; j < 4; ++j) {
    int d = h * DH_ + j * 16 + l16;
#pragma unroll
    for (int r = 0; r < 8; ++r) {
      int m = q0 + r + 8 * half;
      out[(size_t)(b * S_ + m) * E_ + d] = f2bf(o[j][r] * inv[r]);
    }
  }
}

// ---------------------------------------------------------------------------
// Host-side orchestration
// ---------------------------------------------------------------------------
extern "C" void kernel_launch(void* const* d_in, const int* in_sizes, int n_in,
                              void* d_out, int out_size, void* d_ws,
                              size_t ws_size, hipStream_t stream) {
  const int*   indices   = (const int*)d_in[0];
  // d_in[1] attention_mask: unused (block-causal mask computed analytically)
  const float* code_emb  = (const float*)d_in[2];
  const float* stage_emb = (const float*)d_in[3];
  const float* time_pos  = (const float*)d_in[4];
  const float* joint_pos = (const float*)d_in[5];
  const float* in_proj_w = (const float*)d_in[6];
  const float* in_proj_b = (const float*)d_in[7];
  const float* out_w     = (const float*)d_in[8];
  const float* out_b     = (const float*)d_in[9];
  const float* ln1_g     = (const float*)d_in[10];
  const float* ln1_b     = (const float*)d_in[11];
  const float* lin1_w    = (const float*)d_in[12];
  const float* lin1_b    = (const float*)d_in[13];
  const float* lin2_w    = (const float*)d_in[14];
  const float* lin2_b    = (const float*)d_in[15];
  const float* ln2_g     = (const float*)d_in[16];
  const float* ln2_b     = (const float*)d_in[17];

  char* p = (char*)d_ws;
  auto alloc = [&](size_t bytes) -> void* {
    void* r = (void*)p;
    p += (bytes + 255) & ~(size_t)255;
    return r;
  };

  const int nWqkv = L_ * 3 * E_ * E_;
  const int nWout = L_ * E_ * E_;
  const int nW1   = L_ * DFF_ * E_;
  const int nW2   = L_ * E_ * DFF_;

  __bf16* Wqkv = (__bf16*)alloc((size_t)nWqkv * 2);
  __bf16* Wout = (__bf16*)alloc((size_t)nWout * 2);
  __bf16* W1   = (__bf16*)alloc((size_t)nW1 * 2);
  __bf16* W2   = (__bf16*)alloc((size_t)nW2 * 2);
  float*  X    = (float*)alloc((size_t)MTOK * E_ * 4);      // residual stream
  __bf16* Y    = (__bf16*)alloc((size_t)MTOK * E_ * 2);     // LN output
  __bf16* QK   = (__bf16*)alloc((size_t)MTOK * LQK * 2);    // q|k, stride 1024
  __bf16* Vt   = (__bf16*)alloc((size_t)B_ * H_ * DH_ * S_ * 2); // v transposed
  __bf16* AO   = (__bf16*)alloc((size_t)MTOK * E_ * 2);     // attn out
  __bf16* Hb   = (__bf16*)alloc((size_t)MTOK * DFF_ * 2);   // FFN hidden

  auto cvt = [&](const float* s, __bf16* d, int n) {
    cvt_kernel<<<(n + 255) / 256, 256, 0, stream>>>(s, d, n);
  };
  cvt(in_proj_w, Wqkv, nWqkv);
  cvt(out_w,     Wout, nWout);
  cvt(lin1_w,    W1,   nW1);
  cvt(lin2_w,    W2,   nW2);

  embed_kernel<<<MTOK, 128, 0, stream>>>(indices, code_emb, stage_emb, time_pos,
                                         joint_pos, X);

  for (int l = 0; l < L_; ++l) {
    // --- attention sublayer ---
    ln_kernel<<<MTOK, 128, 0, stream>>>(X, ln1_g + l * E_, ln1_b + l * E_, Y);
    gemm_kernel<3><<<dim3(3 * E_ / 256, MTOK / 128), 256, 0, stream>>>(
        Y, Wqkv + (size_t)l * 3 * E_ * E_, in_proj_b + l * 3 * E_, nullptr, QK,
        Vt, MTOK, 3 * E_, E_);
    attn_kernel<<<dim3(S_ / 128, B_ * H_), 256, 0, stream>>>(QK, Vt, AO);
    gemm_kernel<2><<<dim3(E_ / 256, MTOK / 128), 256, 0, stream>>>(
        AO, Wout + (size_t)l * E_ * E_, out_b + l * E_, X, nullptr, nullptr,
        MTOK, E_, E_);

    // --- FFN sublayer ---
    ln_kernel<<<MTOK, 128, 0, stream>>>(X, ln2_g + l * E_, ln2_b + l * E_, Y);
    gemm_kernel<1><<<dim3(DFF_ / 256, MTOK / 128), 256, 0, stream>>>(
        Y, W1 + (size_t)l * DFF_ * E_, lin1_b + l * DFF_, nullptr, Hb, nullptr,
        MTOK, DFF_, E_);
    gemm_kernel<2><<<dim3(E_ / 256, MTOK / 128), 256, 0, stream>>>(
        Hb, W2 + (size_t)l * E_ * DFF_, lin2_b + l * E_, X, nullptr, nullptr,
        MTOK, E_, DFF_);
  }

  hipMemcpyAsync(d_out, X, (size_t)out_size * sizeof(float),
                 hipMemcpyDeviceToDevice, stream);
}
